// Gemma4TextBlock_53601191854595
// MI455X (gfx1250) — compile-verified, matrix operand order checked by reading, not run on hardware
//
#include <hip/hip_runtime.h>

// ---------------- problem constants ----------------
#define BB   4
#define SS   1024
#define DD   2048
#define HH   16
#define KVHH 8
#define HDD  128
#define FFF  8192
#define ROWS (BB * SS)          // 4096 token rows

typedef __attribute__((ext_vector_type(16))) __bf16 v16bf;
typedef __attribute__((ext_vector_type(8)))  __bf16 v8bf;
typedef __attribute__((ext_vector_type(8)))  float  v8f;
typedef __attribute__((ext_vector_type(4)))  int    v4i;

// ---------------- CDNA5 async copy (memory -> LDS, ASYNCcnt) ----------------
#if defined(__gfx1250__) && __has_builtin(__builtin_amdgcn_global_load_async_to_lds_b128)
#define HAVE_ASYNC 1
#else
#define HAVE_ASYNC 0
#endif

__device__ __forceinline__ void async_cp16(const unsigned short* g, unsigned short* l) {
#if HAVE_ASYNC
  __builtin_amdgcn_global_load_async_to_lds_b128(
      (__attribute__((address_space(1))) v4i*)(g),
      (__attribute__((address_space(3))) v4i*)(l), 0, 0);
#else
  (void)g; (void)l;
#endif
}

__device__ __forceinline__ void wait_async0() {
#if defined(__gfx1250__)
#if __has_builtin(__builtin_amdgcn_s_wait_asynccnt)
  __builtin_amdgcn_s_wait_asynccnt(0);
#else
  asm volatile("s_wait_asynccnt 0" ::: "memory");
#endif
#endif
}

// ---------------- helpers ----------------
__device__ __forceinline__ unsigned short f2bf(float f) {
  unsigned int u = __float_as_uint(f);
  u += 0x7fffu + ((u >> 16) & 1u);   // round-to-nearest-even
  return (unsigned short)(u >> 16);
}

// Load one 16-bit A/B WMMA fragment (ISA 7.12.2 layout) given a pointer to
// this lane's row at its k-chunk base. lane: row=(L&15), hi=(L>>4).
// elements 0..7  <- k = hi*8 + e ;  elements 8..15 <- k = 16 + hi*8 + (e-8)
__device__ __forceinline__ v16bf frag_k(const unsigned short* p, int hi) {
  union { v16bf v; v8bf h[2]; } u;
  u.h[0] = *reinterpret_cast<const v8bf*>(p + hi * 8);
  u.h[1] = *reinterpret_cast<const v8bf*>(p + 16 + hi * 8);
  return u.v;
}

__device__ __forceinline__ v8f wmma_bf16(v16bf a, v16bf b, v8f c) {
  return __builtin_amdgcn_wmma_f32_16x16x32_bf16(false, a, false, b, (short)0, c,
                                                 false, false);
}

// block-wide sum for 256-thread blocks (8 waves)
__device__ __forceinline__ float block_sum_256(float v, float* rbuf) {
  #pragma unroll
  for (int off = 16; off; off >>= 1) v += __shfl_xor(v, off, 32);
  int lane = threadIdx.x & 31, w = threadIdx.x >> 5;
  __syncthreads();
  if (lane == 0) rbuf[w] = v;
  __syncthreads();
  float s = 0.f;
  #pragma unroll
  for (int i = 0; i < 8; i++) s += rbuf[i];
  return s;
}

// ---------------- f32 -> bf16 convert (packed b64 stores) ----------------
__global__ __launch_bounds__(256) void cvt_bf16_k(const float* __restrict__ x,
                                                  unsigned short* __restrict__ y,
                                                  int n) {
  int i = (blockIdx.x * 256 + threadIdx.x) * 4;   // n is a multiple of 1024
  if (i + 3 < n) {
    float4 v = *reinterpret_cast<const float4*>(x + i);
    unsigned long long p = (unsigned long long)f2bf(v.x)
                         | ((unsigned long long)f2bf(v.y) << 16)
                         | ((unsigned long long)f2bf(v.z) << 32)
                         | ((unsigned long long)f2bf(v.w) << 48);
    *reinterpret_cast<unsigned long long*>(y + i) = p;
  }
}

// ---------------- RMSNorm over a 2048-wide row -> bf16 ----------------
__global__ __launch_bounds__(256) void rms_row_k(const float* __restrict__ X,
                                                 const float* __restrict__ scale,
                                                 unsigned short* __restrict__ O) {
  __shared__ float rbuf[8];
  int row = blockIdx.x, t = threadIdx.x;
  const float* x = X + (size_t)row * DD;
  float v[8], ss = 0.f;
  #pragma unroll
  for (int i = 0; i < 8; i++) { v[i] = x[t + i * 256]; ss += v[i] * v[i]; }
  float tot = block_sum_256(ss, rbuf);
  float r = rsqrtf(tot * (1.0f / DD) + 1e-6f);
  unsigned short* o = O + (size_t)row * DD;
  #pragma unroll
  for (int i = 0; i < 8; i++) {
    int c = t + i * 256;
    o[c] = f2bf(v[i] * r * scale[c]);
  }
}

// ---------------- TN bf16 GEMM: C[M,N] = A[M,K] * Bw[N,K]^T ----------------
// block tile 128x128, 8 waves of 64x32, K-step 64, double-buffered LDS with
// CDNA5 async global->LDS copies (ASYNCcnt), f32 accumulation.
__global__ __launch_bounds__(256) void gemm_tn_bf16(
    const unsigned short* __restrict__ A, const unsigned short* __restrict__ Bw,
    float* __restrict__ C, int M, int N, int K) {
  constexpr int PITCH = 72;                 // halfs per LDS row (144B, 16B aligned)
  __shared__ unsigned short sA[2][128 * PITCH];
  __shared__ unsigned short sB[2][128 * PITCH];

  int t = threadIdx.x;
  int lane = t & 31, wave = t >> 5;
  int hi = lane >> 4, l15 = lane & 15;
  int m0 = blockIdx.y * 128, n0 = blockIdx.x * 128;
  int wmBase = (wave >> 2) * 64;            // 2x4 wave grid, wave tile 64x32
  int wnBase = (wave & 3) * 32;

  int lrow = t >> 1;                        // 128 rows, 2 threads/row
  int lcol = (t & 1) * 32;                  // halfs [0,32) / [32,64)

  const unsigned short* gA = A + (size_t)(m0 + lrow) * K + lcol;
  const unsigned short* gB = Bw + (size_t)(n0 + lrow) * K + lcol;

  v8f acc[4][2] = {};
  int nsteps = K >> 6;

#if HAVE_ASYNC
  auto issue = [&](int k0, int buf) {
    unsigned short* la = &sA[buf][lrow * PITCH + lcol];
    unsigned short* lb = &sB[buf][lrow * PITCH + lcol];
    #pragma unroll
    for (int c = 0; c < 4; c++) {
      async_cp16(gA + k0 + c * 8, la + c * 8);
      async_cp16(gB + k0 + c * 8, lb + c * 8);
    }
  };
  issue(0, 0);
  for (int s = 0; s < nsteps; s++) {
    int buf = s & 1;
    wait_async0();            // this wave's copies for buf are done
    __syncthreads();          // everyone's copies done; prev compute reads done
    if (s + 1 < nsteps) issue((s + 1) << 6, buf ^ 1);
    #pragma unroll
    for (int kc = 0; kc < 2; kc++) {
      v16bf af[4], bf2[2];
      #pragma unroll
      for (int i = 0; i < 4; i++)
        af[i] = frag_k(&sA[buf][(wmBase + i * 16 + l15) * PITCH + kc * 32], hi);
      #pragma unroll
      for (int j = 0; j < 2; j++)
        bf2[j] = frag_k(&sB[buf][(wnBase + j * 16 + l15) * PITCH + kc * 32], hi);
      #pragma unroll
      for (int i = 0; i < 4; i++)
        #pragma unroll
        for (int j = 0; j < 2; j++)
          acc[i][j] = wmma_bf16(af[i], bf2[j], acc[i][j]);
    }
  }
#else
  for (int s = 0; s < nsteps; s++) {
    int buf = s & 1;
    int k0 = s << 6;
    uint4 ra[4], rb[4];
    #pragma unroll
    for (int c = 0; c < 4; c++) {
      ra[c] = *reinterpret_cast<const uint4*>(gA + k0 + c * 8);
      rb[c] = *reinterpret_cast<const uint4*>(gB + k0 + c * 8);
    }
    __syncthreads();
    #pragma unroll
    for (int c = 0; c < 4; c++) {
      *reinterpret_cast<uint4*>(&sA[buf][lrow * PITCH + lcol + c * 8]) = ra[c];
      *reinterpret_cast<uint4*>(&sB[buf][lrow * PITCH + lcol + c * 8]) = rb[c];
    }
    __syncthreads();
    #pragma unroll
    for (int kc = 0; kc < 2; kc++) {
      v16bf af[4], bf2[2];
      #pragma unroll
      for (int i = 0; i < 4; i++)
        af[i] = frag_k(&sA[buf][(wmBase + i * 16 + l15) * PITCH + kc * 32], hi);
      #pragma unroll
      for (int j = 0; j < 2; j++)
        bf2[j] = frag_k(&sB[buf][(wnBase + j * 16 + l15) * PITCH + kc * 32], hi);
      #pragma unroll
      for (int i = 0; i < 4; i++)
        #pragma unroll
        for (int j = 0; j < 2; j++)
          acc[i][j] = wmma_bf16(af[i], bf2[j], acc[i][j]);
    }
  }
#endif

  // epilogue: row-stepped pointers (no per-store 64-bit multiplies)
  #pragma unroll
  for (int i = 0; i < 4; i++)
    #pragma unroll
    for (int j = 0; j < 2; j++) {
      float* cp = C + (size_t)(m0 + wmBase + i * 16 + hi * 8) * N +
                  (n0 + wnBase + j * 16 + l15);
      #pragma unroll
      for (int r = 0; r < 8; r++) { *cp = acc[i][j][r]; cp += N; }
    }
}

// ---------------- per-head RMS + RoPE -> bf16 [B,nH,S,HD] ----------------
__global__ __launch_bounds__(256) void rms_rope_k(const float* __restrict__ X,
                                                  const float* __restrict__ scale,
                                                  const int* __restrict__ pos,
                                                  unsigned short* __restrict__ O,
                                                  int nH) {
  int lane = threadIdx.x & 31, wave = threadIdx.x >> 5;
  int item = blockIdx.x * 8 + wave;
  int h = item % nH;
  int s = (item / nH) % SS;
  int b = item / (nH * SS);
  const float* x = X + ((size_t)(b * SS + s) * nH + h) * HDD;
  float v0 = x[lane], v1 = x[lane + 32], v2 = x[lane + 64], v3 = x[lane + 96];
  float ss = v0 * v0 + v1 * v1 + v2 * v2 + v3 * v3;
  #pragma unroll
  for (int off = 16; off; off >>= 1) ss += __shfl_xor(ss, off, 32);
  float r = rsqrtf(ss * (1.0f / HDD) + 1e-6f);
  v0 *= r * scale[lane];      v1 *= r * scale[lane + 32];
  v2 *= r * scale[lane + 64]; v3 *= r * scale[lane + 96];
  float p = (float)pos[b * SS + s];
  float f0 = __powf(10000.0f, -(float)lane * (1.0f / 64.0f));
  float f1 = __powf(10000.0f, -(float)(lane + 32) * (1.0f / 64.0f));
  float c0 = __cosf(p * f0), s0 = __sinf(p * f0);
  float c1 = __cosf(p * f1), s1 = __sinf(p * f1);
  unsigned short* o = O + ((size_t)(b * nH + h) * SS + s) * HDD;
  o[lane]      = f2bf(v0 * c0 - v2 * s0);
  o[lane + 64] = f2bf(v2 * c0 + v0 * s0);
  o[lane + 32] = f2bf(v1 * c1 - v3 * s1);
  o[lane + 96] = f2bf(v3 * c1 + v1 * s1);
}

// ---------------- V: RMS (no scale) -> transposed bf16 [B,KVH,HD,S] --------
__global__ __launch_bounds__(256) void v_rms_t_k(const float* __restrict__ X,
                                                 unsigned short* __restrict__ O) {
  int lane = threadIdx.x & 31, wave = threadIdx.x >> 5;
  int item = blockIdx.x * 8 + wave;
  int h = item % KVHH;
  int s = (item / KVHH) % SS;
  int b = item / (KVHH * SS);
  const float* x = X + ((size_t)(b * SS + s) * KVHH + h) * HDD;
  float v[4];
  float ss = 0.f;
  #pragma unroll
  for (int i = 0; i < 4; i++) { v[i] = x[lane + i * 32]; ss += v[i] * v[i]; }
  #pragma unroll
  for (int off = 16; off; off >>= 1) ss += __shfl_xor(ss, off, 32);
  float r = rsqrtf(ss * (1.0f / HDD) + 1e-6f);
  unsigned short* o = O + ((size_t)(b * KVHH + h) * HDD) * SS;
  #pragma unroll
  for (int i = 0; i < 4; i++) o[(size_t)(lane + i * 32) * SS + s] = f2bf(v[i] * r);
}

// ---------------- flash attention (causal, softcap 50) ----------------
// grid (S/64, H, B); 128 threads = 4 waves; each wave: 16 q-rows
__global__ __launch_bounds__(128) void flash_attn_k(
    const unsigned short* __restrict__ Qb,   // [B,H,S,HD]
    const unsigned short* __restrict__ Kb,   // [B,KVH,S,HD]
    const unsigned short* __restrict__ Vt,   // [B,KVH,HD,S]
    unsigned short* __restrict__ Ob) {       // [B,S,H,HD]
  __shared__ unsigned short sP[4][16 * 40];  // per-wave 16x32 P tile, pitch 40
  int t = threadIdx.x, lane = t & 31, wave = t >> 5;
  int hi = lane >> 4, l15 = lane & 15;
  int qTile = blockIdx.x, h = blockIdx.y, b = blockIdx.z;
  int kvh = h >> 1;                          // groups = H/KVH = 2
  int qRow0 = qTile * 64 + wave * 16;

  const unsigned short* Q = Qb + ((size_t)(b * HH + h) * SS) * HDD;
  const unsigned short* Kp = Kb + ((size_t)(b * KVHH + kvh) * SS) * HDD;
  const unsigned short* Vp = Vt + ((size_t)(b * KVHH + kvh) * HDD) * SS;

  v16bf qf[4];
  #pragma unroll
  for (int kk = 0; kk < 4; kk++)
    qf[kk] = frag_k(Q + (size_t)(qRow0 + l15) * HDD + kk * 32, hi);

  v8f o[8] = {};
  float rmax[8], rsum[8];
  #pragma unroll
  for (int r = 0; r < 8; r++) { rmax[r] = -1e30f; rsum[r] = 0.f; }

  int kEnd = (qTile + 1) * 64;
  for (int kt = 0; kt < kEnd; kt += 32) {
    // ---- scores: Q(16x128) x K^T -> two 16x16 tiles
    v8f sc[2] = {};
    #pragma unroll
    for (int j = 0; j < 2; j++) {
      const unsigned short* krow = Kp + (size_t)(kt + j * 16 + l15) * HDD;
      #pragma unroll
      for (int kk = 0; kk < 4; kk++) {
        v16bf kf = frag_k(krow + kk * 32, hi);
        sc[j] = wmma_bf16(qf[kk], kf, sc[j]);
      }
    }
    // ---- softcap + causal mask + row max
    float newmax[8];
    #pragma unroll
    for (int r = 0; r < 8; r++) {
      int qpos = qRow0 + hi * 8 + r;
      float mx = rmax[r];
      #pragma unroll
      for (int j = 0; j < 2; j++) {
        int kpos = kt + j * 16 + l15;
        float s = sc[j][r];
        s = tanhf(s * 0.02f) * 50.0f;
        if (kpos > qpos) s = -1e30f;
        sc[j][r] = s;
        mx = fmaxf(mx, s);
      }
      #pragma unroll
      for (int off = 1; off < 16; off <<= 1) mx = fmaxf(mx, __shfl_xor(mx, off, 32));
      newmax[r] = mx;
    }
    // ---- rescale running state, exponentiate, accumulate row sums
    #pragma unroll
    for (int r = 0; r < 8; r++) {
      float corr = __expf(rmax[r] - newmax[r]);
      rmax[r] = newmax[r];
      rsum[r] *= corr;
      #pragma unroll
      for (int d = 0; d < 8; d++) o[d][r] *= corr;
      float p0 = __expf(sc[0][r] - newmax[r]);
      float p1 = __expf(sc[1][r] - newmax[r]);
      float ps = p0 + p1;
      #pragma unroll
      for (int off = 1; off < 16; off <<= 1) ps += __shfl_xor(ps, off, 32);
      rsum[r] += ps;
      sP[wave][(hi * 8 + r) * 40 + l15] = f2bf(p0);
      sP[wave][(hi * 8 + r) * 40 + 16 + l15] = f2bf(p1);
    }
    // ---- P(16x32) x V(32x16 per hd-subtile)
    v16bf pf = frag_k(&sP[wave][l15 * 40], hi);
    #pragma unroll
    for (int d = 0; d < 8; d++) {
      const unsigned short* vrow = Vp + (size_t)(d * 16 + l15) * SS + kt;
      v16bf vf = frag_k(vrow, hi);
      o[d] = wmma_bf16(pf, vf, o[d]);
    }
  }
  // ---- normalize + write [B,S,H,HD]
  #pragma unroll
  for (int r = 0; r < 8; r++) {
    float inv = 1.0f / rsum[r];
    int srow = qRow0 + hi * 8 + r;
    unsigned short* orow = Ob + ((size_t)(b * SS + srow) * HH + h) * HDD;
    #pragma unroll
    for (int d = 0; d < 8; d++) orow[d * 16 + l15] = f2bf(o[d][r] * inv);
  }
}

// ---------------- residual + post-attn RMS + pre-ffn RMS ----------------
__global__ __launch_bounds__(256) void attn_resid_k(
    const float* __restrict__ x, const float* __restrict__ attnout,
    const float* __restrict__ post_scale, const float* __restrict__ preffn_scale,
    float* __restrict__ xnew, unsigned short* __restrict__ h2) {
  __shared__ float rbuf[8];
  int row = blockIdx.x, t = threadIdx.x;
  const float* a = attnout + (size_t)row * DD;
  const float* xr = x + (size_t)row * DD;
  float av[8], ss = 0.f;
  #pragma unroll
  for (int i = 0; i < 8; i++) { av[i] = a[t + i * 256]; ss += av[i] * av[i]; }
  float r1 = rsqrtf(block_sum_256(ss, rbuf) * (1.0f / DD) + 1e-6f);
  float xv[8];
  float ss2 = 0.f;
  #pragma unroll
  for (int i = 0; i < 8; i++) {
    int c = t + i * 256;
    float xn = xr[c] + av[i] * r1 * post_scale[c];
    xv[i] = xn;
    ss2 += xn * xn;
    xnew[(size_t)row * DD + c] = xn;
  }
  float r2 = rsqrtf(block_sum_256(ss2, rbuf) * (1.0f / DD) + 1e-6f);
  #pragma unroll
  for (int i = 0; i < 8; i++) {
    int c = t + i * 256;
    h2[(size_t)row * DD + c] = f2bf(xv[i] * r2 * preffn_scale[c]);
  }
}

// ---------------- SwiGLU: act = gelu_tanh(gate) * up -> bf16 (b64 stores) ---
__global__ __launch_bounds__(256) void swiglu_k(const float* __restrict__ g,
                                                const float* __restrict__ u,
                                                unsigned short* __restrict__ act) {
  int i = (blockIdx.x * 256 + threadIdx.x) * 4;
  float4 gv = *reinterpret_cast<const float4*>(g + i);
  float4 uv = *reinterpret_cast<const float4*>(u + i);
  auto gelu = [](float x) {
    return 0.5f * x * (1.0f + tanhf(0.7978845608f * (x + 0.044715f * x * x * x)));
  };
  unsigned long long p = (unsigned long long)f2bf(gelu(gv.x) * uv.x)
                       | ((unsigned long long)f2bf(gelu(gv.y) * uv.y) << 16)
                       | ((unsigned long long)f2bf(gelu(gv.z) * uv.z) << 32)
                       | ((unsigned long long)f2bf(gelu(gv.w) * uv.w) << 48);
  *reinterpret_cast<unsigned long long*>(act + i) = p;
}

// ---------------- final: out = (xnew + rms(ffn,post_ffn)) * layer_scalar ----
__global__ __launch_bounds__(256) void final_k(const float* __restrict__ xnew,
                                               const float* __restrict__ ffn,
                                               const float* __restrict__ post_scale,
                                               const float* __restrict__ lscal,
                                               float* __restrict__ out) {
  __shared__ float rbuf[8];
  int row = blockIdx.x, t = threadIdx.x;
  const float* f = ffn + (size_t)row * DD;
  float fv[8], ss = 0.f;
  #pragma unroll
  for (int i = 0; i < 8; i++) { fv[i] = f[t + i * 256]; ss += fv[i] * fv[i]; }
  float r = rsqrtf(block_sum_256(ss, rbuf) * (1.0f / DD) + 1e-6f);
  float ls = lscal[0];
  #pragma unroll
  for (int i = 0; i < 8; i++) {
    int c = t + i * 256;
    out[(size_t)row * DD + c] =
        (xnew[(size_t)row * DD + c] + fv[i] * r * post_scale[c]) * ls;
  }
}

// ---------------- launch ----------------
extern "C" void kernel_launch(void* const* d_in, const int* in_sizes, int n_in,
                              void* d_out, int out_size, void* d_ws, size_t ws_size,
                              hipStream_t stream) {
  const float* x        = (const float*)d_in[0];
  const int*   pos      = (const int*)d_in[1];
  // d_in[2] attention_mask (causal, assumed), d_in[3] query_mask (all ones): unused
  const float* wq       = (const float*)d_in[4];
  const float* wk       = (const float*)d_in[5];
  const float* wv       = (const float*)d_in[6];
  const float* wo       = (const float*)d_in[7];
  const float* q_scale  = (const float*)d_in[8];
  const float* k_scale  = (const float*)d_in[9];
  const float* pre_attn = (const float*)d_in[10];
  const float* post_attn= (const float*)d_in[11];
  const float* pre_ffn  = (const float*)d_in[12];
  const float* post_ffn = (const float*)d_in[13];
  const float* wg       = (const float*)d_in[14];
  const float* wu       = (const float*)d_in[15];
  const float* wd       = (const float*)d_in[16];
  const float* lscal    = (const float*)d_in[17];
  float* out = (float*)d_out;

  char* ws = (char*)d_ws;
  size_t off = 0;
  auto take = [&](size_t bytes) {
    char* p = ws + off;
    off += (bytes + 255) & ~(size_t)255;
    return p;
  };

  // bf16 weights (recomputed each call; deterministic)
  unsigned short* wq_b = (unsigned short*)take((size_t)2048 * 2048 * 2);
  unsigned short* wk_b = (unsigned short*)take((size_t)1024 * 2048 * 2);
  unsigned short* wv_b = (unsigned short*)take((size_t)1024 * 2048 * 2);
  unsigned short* wo_b = (unsigned short*)take((size_t)2048 * 2048 * 2);
  unsigned short* wg_b = (unsigned short*)take((size_t)8192 * 2048 * 2);
  unsigned short* wu_b = (unsigned short*)take((size_t)8192 * 2048 * 2);
  unsigned short* wd_b = (unsigned short*)take((size_t)2048 * 8192 * 2);
  // activations
  unsigned short* h_b  = (unsigned short*)take((size_t)ROWS * DD * 2);  // reused: attn_b
  unsigned short* q_b  = (unsigned short*)take((size_t)ROWS * 2048 * 2);
  unsigned short* k_b  = (unsigned short*)take((size_t)ROWS * 1024 * 2);
  unsigned short* v_bt = (unsigned short*)take((size_t)ROWS * 1024 * 2);
  float* qf32 = (float*)take((size_t)ROWS * 2048 * 4);   // reused: attnout
  float* kf32 = (float*)take((size_t)ROWS * 1024 * 4);   // kf32+vf32 reused: ffn
  float* vf32 = (float*)take((size_t)ROWS * 1024 * 4);
  float* xnew = (float*)take((size_t)ROWS * DD * 4);
  unsigned short* h2_b = (unsigned short*)take((size_t)ROWS * DD * 2);
  float* gate = (float*)take((size_t)ROWS * FFF * 4);
  float* up   = (float*)take((size_t)ROWS * FFF * 4);
  unsigned short* act_b = (unsigned short*)take((size_t)ROWS * FFF * 2);

  unsigned short* attn_b  = h_b;    // h dead after QKV projections
  float*          attnout = qf32;   // q f32 dead after rope
  float*          ffn     = kf32;   // k/v f32 dead after rope (32MB contiguous)

  dim3 blk256(256), blk128(128);

  // 1) weight conversion
  cvt_bf16_k<<<(2048 * 2048) / 1024, blk256, 0, stream>>>(wq, wq_b, 2048 * 2048);
  cvt_bf16_k<<<(1024 * 2048) / 1024, blk256, 0, stream>>>(wk, wk_b, 1024 * 2048);
  cvt_bf16_k<<<(1024 * 2048) / 1024, blk256, 0, stream>>>(wv, wv_b, 1024 * 2048);
  cvt_bf16_k<<<(2048 * 2048) / 1024, blk256, 0, stream>>>(wo, wo_b, 2048 * 2048);
  cvt_bf16_k<<<(8192 * 2048) / 1024, blk256, 0, stream>>>(wg, wg_b, 8192 * 2048);
  cvt_bf16_k<<<(8192 * 2048) / 1024, blk256, 0, stream>>>(wu, wu_b, 8192 * 2048);
  cvt_bf16_k<<<(2048 * 8192) / 1024, blk256, 0, stream>>>(wd, wd_b, 2048 * 8192);

  // 2) pre-attn RMSNorm -> h_b (bf16)
  rms_row_k<<<ROWS, blk256, 0, stream>>>(x, pre_attn, h_b);

  // 3) QKV projections
  gemm_tn_bf16<<<dim3(2048 / 128, ROWS / 128), blk256, 0, stream>>>(
      h_b, wq_b, qf32, ROWS, 2048, 2048);
  gemm_tn_bf16<<<dim3(1024 / 128, ROWS / 128), blk256, 0, stream>>>(
      h_b, wk_b, kf32, ROWS, 1024, 2048);
  gemm_tn_bf16<<<dim3(1024 / 128, ROWS / 128), blk256, 0, stream>>>(
      h_b, wv_b, vf32, ROWS, 1024, 2048);

  // 4) per-head RMS + RoPE (q,k) ; RMS + transpose (v)
  rms_rope_k<<<(BB * SS * HH) / 8, blk256, 0, stream>>>(qf32, q_scale, pos, q_b, HH);
  rms_rope_k<<<(BB * SS * KVHH) / 8, blk256, 0, stream>>>(kf32, k_scale, pos, k_b, KVHH);
  v_rms_t_k<<<(BB * SS * KVHH) / 8, blk256, 0, stream>>>(vf32, v_bt);

  // 5) flash attention -> attn_b [B,S,H,HD]
  flash_attn_k<<<dim3(SS / 64, HH, BB), blk128, 0, stream>>>(q_b, k_b, v_bt, attn_b);

  // 6) output projection
  gemm_tn_bf16<<<dim3(2048 / 128, ROWS / 128), blk256, 0, stream>>>(
      attn_b, wo_b, attnout, ROWS, 2048, 2048);

  // 7) residual + post-attn RMS + pre-ffn RMS
  attn_resid_k<<<ROWS, blk256, 0, stream>>>(x, attnout, post_attn, pre_ffn, xnew, h2_b);

  // 8) FFN
  gemm_tn_bf16<<<dim3(FFF / 128, ROWS / 128), blk256, 0, stream>>>(
      h2_b, wg_b, gate, ROWS, FFF, 2048);
  gemm_tn_bf16<<<dim3(FFF / 128, ROWS / 128), blk256, 0, stream>>>(
      h2_b, wu_b, up, ROWS, FFF, 2048);
  swiglu_k<<<(ROWS * FFF) / 1024, blk256, 0, stream>>>(gate, up, act_b);
  gemm_tn_bf16<<<dim3(2048 / 128, ROWS / 128), blk256, 0, stream>>>(
      act_b, wd_b, ffn, ROWS, 2048, FFF);

  // 9) final residual + post-ffn RMS + layer scalar
  final_k<<<ROWS, blk256, 0, stream>>>(xnew, ffn, post_ffn, lscal, out);
}
